// RNN_82566451298754
// MI455X (gfx1250) — compile-verified, matrix operand order checked by reading, not run on hardware
//
#include <hip/hip_runtime.h>

#define T_STEPS 512
#define BATCH   32
#define DIM     1024
#define HID     1024
#define G3      3072       // 3*HID
#define KT_N    32         // DIM/32 k-tiles
#define CTL_N   192        // G3/16 col-tiles

typedef __attribute__((ext_vector_type(16))) __bf16         bf16x16;
typedef __attribute__((ext_vector_type(16))) unsigned short u16x16;
typedef __attribute__((ext_vector_type(8)))  float          f32x8;
typedef __attribute__((ext_vector_type(4)))  unsigned int   u32x4;

union V16 {
    u16x16 v;
    u32x4 q[2];
    unsigned short s[16];
};

__device__ __forceinline__ unsigned short bf16_bits(float f) {
    unsigned u = __float_as_uint(f);
    unsigned r = u + 0x7FFFu + ((u >> 16) & 1u);   // round-to-nearest-even
    return (unsigned short)(r >> 16);
}
__device__ __forceinline__ float bits_to_f32(unsigned short s) {
    return __uint_as_float(((unsigned)s) << 16);
}
// wave32 16-bit A-matrix 16x32 K pattern (ISA 7.12.2): lane<16 holds K {0..7,16..23},
// lane>=16 holds K {8..15,24..31}; j is the half index 0..15 within the 8-VGPR operand.
__device__ __forceinline__ int koff(int lane, int j) {
    return (j < 8 ? j : j + 8) + ((lane & 16) ? 8 : 0);
}
__device__ __forceinline__ u16x16 load_v16(const unsigned short* p) {
    V16 t;
    t.q[0] = *(const u32x4*)(p);
    t.q[1] = *(const u32x4*)(p + 8);
    return t.v;
}
__device__ __forceinline__ f32x8 wmma_bf16(u16x16 a, u16x16 b, f32x8 c) {
    return __builtin_amdgcn_wmma_f32_16x16x32_bf16(
        false, __builtin_bit_cast(bf16x16, a),
        false, __builtin_bit_cast(bf16x16, b),
        (short)0, c, false, false);
}

// ---- Pre-split + pre-swizzle W (f32 [1024 x 3072]) into WMMA-B-layout bf16 hi/lo ----
// Packed index: ((kt*CTL_N + ct)*32 + lane)*16 + j ; each lane owns 32 contiguous bytes.
__global__ void prep_w(const float* __restrict__ W,
                       unsigned short* __restrict__ Whi,
                       unsigned short* __restrict__ Wlo) {
    int idx = blockIdx.x * blockDim.x + threadIdx.x;
    if (idx >= KT_N * CTL_N * 32) return;
    int lane = idx & 31;
    int tile = idx >> 5;
    int ct = tile % CTL_N;
    int kt = tile / CTL_N;
    int col = ct * 16 + (lane & 15);
    unsigned short* hi = Whi + (size_t)idx * 16;
    unsigned short* lo = Wlo + (size_t)idx * 16;
#pragma unroll
    for (int j = 0; j < 16; ++j) {
        int k = kt * 32 + koff(lane, j);
        float f = W[(size_t)k * G3 + col];
        unsigned short h = bf16_bits(f);
        hi[j] = h;
        lo[j] = bf16_bits(f - bits_to_f32(h));
    }
}

// bias for gi: b_ih everywhere + b_hh folded in for r,z gates only (n gate's b_hh
// is inside the r*(h@W+b_hh) term and must stay with gh).
__global__ void prep_bias(const float* __restrict__ bih, const float* __restrict__ bhh,
                          float* __restrict__ bias) {
    int i = blockIdx.x * blockDim.x + threadIdx.x;
    if (i < G3) bias[i] = bih[i] + (i < 2 * HID ? bhh[i] : 0.0f);
}

// ---- Pack a chunk of X rows into WMMA-A-layout bf16 hi/lo ----
// Packed index: ((rt*KT_N + kt)*32 + lane)*16 + j
__global__ void prep_x(const float* __restrict__ x,
                       unsigned short* __restrict__ Xhi,
                       unsigned short* __restrict__ Xlo, int rows) {
    int idx = blockIdx.x * blockDim.x + threadIdx.x;
    int RT = rows >> 4;
    if (idx >= RT * KT_N * 32) return;
    int lane = idx & 31;
    int tile = idx >> 5;
    int kt = tile % KT_N;
    int rt = tile / KT_N;
    const float* xr = x + (size_t)(rt * 16 + (lane & 15)) * DIM;
    unsigned short* hi = Xhi + (size_t)idx * 16;
    unsigned short* lo = Xlo + (size_t)idx * 16;
#pragma unroll
    for (int j = 0; j < 16; ++j) {
        float f = xr[kt * 32 + koff(lane, j)];
        unsigned short h = bf16_bits(f);
        hi[j] = h;
        lo[j] = bf16_bits(f - bits_to_f32(h));
    }
}

// ---- Big parallel GEMM: gi = X @ W_ih + bias ----
// Each wave owns a 32x64 macro-tile (2 row-tiles x 4 col-tiles): A fragments reused
// 4x, B fragments reused 2x -> 24 b128 loads per 24 WMMAs.
__global__ void __launch_bounds__(256)
gemm_gi(const unsigned short* __restrict__ Xhi, const unsigned short* __restrict__ Xlo,
        const unsigned short* __restrict__ Whi, const unsigned short* __restrict__ Wlo,
        const float* __restrict__ bias, float* __restrict__ gi, int rows) {
    int wave = (int)((blockIdx.x * blockDim.x + threadIdx.x) >> 5);
    int lane = threadIdx.x & 31;
    int RT2 = rows >> 5;                       // row-tile pairs
    const int CG = CTL_N / 4;                  // 48 col groups
    if (wave >= RT2 * CG) return;              // wave-uniform guard (EXEC all-1 for WMMA)
    int cg = wave % CG;
    int rg = wave / CG;
    int rt0 = rg * 2;
    int ct0 = cg * 4;

    const unsigned short* a0h = Xhi + (((size_t)(rt0 + 0) * KT_N) * 32 + lane) * 16;
    const unsigned short* a0l = Xlo + (((size_t)(rt0 + 0) * KT_N) * 32 + lane) * 16;
    const unsigned short* a1h = Xhi + (((size_t)(rt0 + 1) * KT_N) * 32 + lane) * 16;
    const unsigned short* a1l = Xlo + (((size_t)(rt0 + 1) * KT_N) * 32 + lane) * 16;
    const unsigned short* bh = Whi + ((size_t)ct0 * 32 + lane) * 16;
    const unsigned short* bl = Wlo + ((size_t)ct0 * 32 + lane) * 16;

    f32x8 acc[2][4];
#pragma unroll
    for (int r = 0; r < 2; ++r)
#pragma unroll
        for (int c = 0; c < 4; ++c) acc[r][c] = f32x8{};

    for (int kt = 0; kt < KT_N; ++kt) {
        u16x16 A0h = load_v16(a0h), A0l = load_v16(a0l);
        u16x16 A1h = load_v16(a1h), A1l = load_v16(a1l);
#pragma unroll
        for (int c = 0; c < 4; ++c) {
            u16x16 Bh = load_v16(bh + (size_t)c * 32 * 16);
            u16x16 Bl = load_v16(bl + (size_t)c * 32 * 16);
            acc[0][c] = wmma_bf16(A0h, Bh, acc[0][c]);
            acc[0][c] = wmma_bf16(A0h, Bl, acc[0][c]);
            acc[0][c] = wmma_bf16(A0l, Bh, acc[0][c]);
            acc[1][c] = wmma_bf16(A1h, Bh, acc[1][c]);
            acc[1][c] = wmma_bf16(A1h, Bl, acc[1][c]);
            acc[1][c] = wmma_bf16(A1l, Bh, acc[1][c]);
        }
        a0h += 32 * 16; a0l += 32 * 16; a1h += 32 * 16; a1l += 32 * 16;
        bh += (size_t)CTL_N * 32 * 16;  bl += (size_t)CTL_N * 32 * 16;
    }
#pragma unroll
    for (int r = 0; r < 2; ++r) {
        int rbase = (rt0 + r) * 16 + ((lane & 16) ? 8 : 0);  // C layout: VGPR v -> M=v / v+8
#pragma unroll
        for (int c = 0; c < 4; ++c) {
            int col = (ct0 + c) * 16 + (lane & 15);
            float bb = bias[col];
#pragma unroll
            for (int v = 0; v < 8; ++v)
                gi[(size_t)(rbase + v) * G3 + col] = acc[r][c][v] + bb;
        }
    }
}

// ---- One recurrent step ----
// gh = h_prev @ W_hh : A comes pre-packed in WMMA layout (written by the previous
// step), 3 gate tiles, 4-wave K-split + LDS reduction, fused gate math; also emits
// the packed bf16 hi/lo A fragments of h_new for the next step (ping-pong).
__global__ void __launch_bounds__(128)
gru_step(const float* __restrict__ hprev, int first,
         const unsigned short* __restrict__ aHi, const unsigned short* __restrict__ aLo,
         const unsigned short* __restrict__ Whi, const unsigned short* __restrict__ Wlo,
         const float* __restrict__ gi_t, const float* __restrict__ bhh,
         const float* __restrict__ pad_t, float* __restrict__ hout,
         unsigned short* __restrict__ oHi, unsigned short* __restrict__ oLo) {
    __shared__ float red[3][4][256];
    int lane = threadIdx.x & 31;
    int w = threadIdx.x >> 5;                 // wave id 0..3 -> K range
    int rowTile = blockIdx.x & 1;             // 2 x 16 batch rows
    int colTile = blockIdx.x >> 1;            // 64 x 16 hidden cols

    f32x8 acc[3] = {f32x8{}, f32x8{}, f32x8{}};
    if (!first) {
        const unsigned short* ah = aHi + (((size_t)rowTile * KT_N + w * 8) * 32 + lane) * 16;
        const unsigned short* al = aLo + (((size_t)rowTile * KT_N + w * 8) * 32 + lane) * 16;
        for (int i = 0; i < 8; ++i) {
            int kt = w * 8 + i;
            u16x16 Ah = load_v16(ah), Al = load_v16(al);
#pragma unroll
            for (int g = 0; g < 3; ++g) {
                int ct = g * 64 + colTile;
                const unsigned short* bp = Whi + (((size_t)kt * CTL_N + ct) * 32 + lane) * 16;
                const unsigned short* lp = Wlo + (((size_t)kt * CTL_N + ct) * 32 + lane) * 16;
                u16x16 Bh = load_v16(bp), Bl = load_v16(lp);
                acc[g] = wmma_bf16(Ah, Bh, acc[g]);
                acc[g] = wmma_bf16(Ah, Bl, acc[g]);
                acc[g] = wmma_bf16(Al, Bh, acc[g]);
            }
            ah += 32 * 16;
            al += 32 * 16;
        }
    }
#pragma unroll
    for (int g = 0; g < 3; ++g)
#pragma unroll
        for (int v = 0; v < 8; ++v)
            red[g][w][v * 32 + lane] = acc[g][v];
    __syncthreads();

    if (threadIdx.x < 32) {
        int n = lane & 15;                     // local col within tile
        int col = colTile * 16 + n;
        int bbase = rowTile * 16 + ((lane & 16) ? 8 : 0);
        float bn = bhh[2 * HID + col];
        // inverse A-layout mapping for packing h_new: element (m, col) lives at
        // lane m+((col&8)?16:0), half j=(col&7)+((colTile&1)?8:0), k-tile colTile>>1
        int ktp = colTile >> 1;
        int jp = (n & 7) + ((colTile & 1) ? 8 : 0);
        int lp_add = (n & 8) ? 16 : 0;
#pragma unroll
        for (int v = 0; v < 8; ++v) {
            int b = bbase + v;
            float gr = 0.f, gz = 0.f, gn = 0.f;
#pragma unroll
            for (int ww = 0; ww < 4; ++ww) {
                gr += red[0][ww][v * 32 + lane];
                gz += red[1][ww][v * 32 + lane];
                gn += red[2][ww][v * 32 + lane];
            }
            const float* gir = gi_t + (size_t)b * G3;
            float r = 1.f / (1.f + __expf(-(gir[col] + gr)));
            float z = 1.f / (1.f + __expf(-(gir[HID + col] + gz)));
            float nn = tanhf(gir[2 * HID + col] + r * (gn + bn));
            float hp = first ? 0.f : hprev[(size_t)b * HID + col];
            float hn = (1.f - z) * nn + z * hp;
            float p = pad_t[b];
            hn = p * hp + (1.f - p) * hn;
            hout[(size_t)b * HID + col] = hn;
            // packed write for next step's WMMA A operand
            int m = b - rowTile * 16;
            size_t pidx = (((size_t)rowTile * KT_N + ktp) * 32 + (m + lp_add)) * 16 + jp;
            unsigned short h = bf16_bits(hn);
            oHi[pidx] = h;
            oLo[pidx] = bf16_bits(hn - bits_to_f32(h));
        }
    }
}

__global__ void tail_copy(const float* __restrict__ last, float* __restrict__ dst) {
    int i = blockIdx.x * blockDim.x + threadIdx.x;
    if (i < BATCH * HID) {
        float v = last[i];
        dst[i] = v;
        dst[BATCH * HID + i] = v;
    }
}

extern "C" void kernel_launch(void* const* d_in, const int* in_sizes, int n_in,
                              void* d_out, int out_size, void* d_ws, size_t ws_size,
                              hipStream_t stream) {
    const float* x   = (const float*)d_in[0];   // [T,B,D]
    const float* pad = (const float*)d_in[1];   // [T,B,1]
    const float* Wih = (const float*)d_in[2];   // [D,3H]
    const float* Whh = (const float*)d_in[3];   // [H,3H]
    const float* bih = (const float*)d_in[4];   // [3H]
    const float* bhh = (const float*)d_in[5];   // [3H]
    float* out = (float*)d_out;                 // [T,B,H] + [B,H] + [B,H]

    char* ws = (char*)d_ws;
    size_t off = 0;
    auto take = [&](size_t bytes) -> char* {
        char* p = ws + off;
        off = (off + bytes + 255) & ~(size_t)255;
        return p;
    };
    const size_t wElems = (size_t)KT_N * CTL_N * 32 * 16;   // 3,145,728 per array
    unsigned short* WihHi = (unsigned short*)take(wElems * 2);
    unsigned short* WihLo = (unsigned short*)take(wElems * 2);
    unsigned short* WhhHi = (unsigned short*)take(wElems * 2);
    unsigned short* WhhLo = (unsigned short*)take(wElems * 2);
    float* bias = (float*)take(G3 * sizeof(float));
    // ping-pong packed h (WMMA-A layout): 2 row-tiles * 32 kt * 32 lanes * 16 halves
    const size_t hpBytes = (size_t)2 * KT_N * 32 * 16 * 2;  // 64 KB
    unsigned short* hpHi[2] = {(unsigned short*)take(hpBytes), (unsigned short*)take(hpBytes)};
    unsigned short* hpLo[2] = {(unsigned short*)take(hpBytes), (unsigned short*)take(hpBytes)};
    size_t fixed = off;

    // chunk timesteps so (packed X hi/lo + gi) fit in remaining workspace
    const size_t perT = (size_t)BATCH * DIM * 2 * 2 + (size_t)BATCH * G3 * 4 + 1024;
    int CT = T_STEPS;
    if (ws_size > fixed) {
        size_t c = (ws_size - fixed) / perT;
        if (c < (size_t)CT) CT = (int)c;
    } else {
        CT = 1;
    }
    if (CT < 1) CT = 1;
    unsigned short* Xhi = (unsigned short*)take((size_t)CT * BATCH * DIM * 2);
    unsigned short* Xlo = (unsigned short*)take((size_t)CT * BATCH * DIM * 2);
    float* gi = (float*)take((size_t)CT * BATCH * G3 * 4);

    const int wthreads = KT_N * CTL_N * 32;
    prep_w<<<(wthreads + 255) / 256, 256, 0, stream>>>(Wih, WihHi, WihLo);
    prep_w<<<(wthreads + 255) / 256, 256, 0, stream>>>(Whh, WhhHi, WhhLo);
    prep_bias<<<(G3 + 255) / 256, 256, 0, stream>>>(bih, bhh, bias);

    for (int t0 = 0; t0 < T_STEPS; t0 += CT) {
        int ct = (T_STEPS - t0 < CT) ? (T_STEPS - t0) : CT;
        int rows = ct * BATCH;
        int xthreads = (rows / 16) * KT_N * 32;
        prep_x<<<(xthreads + 255) / 256, 256, 0, stream>>>(
            x + (size_t)t0 * BATCH * DIM, Xhi, Xlo, rows);
        int waves = (rows / 32) * (CTL_N / 4);
        gemm_gi<<<(waves + 7) / 8, 256, 0, stream>>>(Xhi, Xlo, WihHi, WihLo, bias, gi, rows);
        for (int tt = 0; tt < ct; ++tt) {
            int t = t0 + tt;
            const float* hprev = (t == 0) ? out : out + (size_t)(t - 1) * BATCH * HID;
            int wr = t & 1;
            int rd = 1 - wr;   // previous step's packed h
            gru_step<<<128, 128, 0, stream>>>(hprev, t == 0 ? 1 : 0,
                                              hpHi[rd], hpLo[rd], WhhHi, WhhLo,
                                              gi + (size_t)tt * BATCH * G3, bhh,
                                              pad + (size_t)t * BATCH,
                                              out + (size_t)t * BATCH * HID,
                                              hpHi[wr], hpLo[wr]);
        }
    }
    tail_copy<<<(BATCH * HID + 255) / 256, 256, 0, stream>>>(
        out + (size_t)(T_STEPS - 1) * BATCH * HID,
        out + (size_t)T_STEPS * BATCH * HID);
}
